// GraphMAEModel_51187420233793
// MI455X (gfx1250) — matrix operand order repaired and checked by reading, not compiled, not run on hardware
//
#include <hip/hip_runtime.h>
#include <hip/hip_bf16.h>

typedef __attribute__((ext_vector_type(16))) _Float16 v16h;
typedef __attribute__((ext_vector_type(8)))  float    v8f;

#define HD 64  // hidden dim, fixed by the model

// ---------------------------------------------------------------------------
// small elementwise kernels
// ---------------------------------------------------------------------------
__global__ void zero_f4_kernel(float4* __restrict__ p, long long n4) {
    long long t = (long long)blockIdx.x * blockDim.x + threadIdx.x;
    if (t < n4) p[t] = make_float4(0.f, 0.f, 0.f, 0.f);
}

__global__ void copy_kernel(const float* __restrict__ src, float* __restrict__ dst, long long n) {
    long long t = (long long)blockIdx.x * blockDim.x + threadIdx.x;
    if (t < n) dst[t] = src[t];
}

// use_x[token_nodes] = enc_mask_token  (token rows become 0 + mask_token)
__global__ void token_set_kernel(float* __restrict__ h, const int* __restrict__ tok,
                                 const float* __restrict__ mask_token, int ntok) {
    long long t = (long long)blockIdx.x * blockDim.x + threadIdx.x;
    if (t >= (long long)ntok * 128) return;
    int i = (int)(t >> 7), c = (int)(t & 127);
    h[(long long)tok[i] * 128 + c] = mask_token[c];
}

// use_x[noise_nodes] = x[noise_src]
__global__ void noise_set_kernel(float* __restrict__ h, const float* __restrict__ x,
                                 const int* __restrict__ nn, const int* __restrict__ ns, int nnoise) {
    long long t = (long long)blockIdx.x * blockDim.x + threadIdx.x;
    if (t >= (long long)nnoise * 128) return;
    int i = (int)(t >> 7), c = (int)(t & 127);
    h[(long long)nn[i] * 128 + c] = x[(long long)ns[i] * 128 + c];
}

// rep = enc @ w_e2d  (100k x 16 @ 16x16; tiny, VALU is fine)
__global__ void e2d_kernel(const float* __restrict__ enc, const float* __restrict__ w,
                           float* __restrict__ rep, int n) {
    long long t = (long long)blockIdx.x * blockDim.x + threadIdx.x;
    if (t >= (long long)n * 16) return;
    int node = (int)(t >> 4), c = (int)(t & 15);
    const float* e = enc + (long long)node * 16;
    float s = 0.f;
#pragma unroll
    for (int k = 0; k < 16; ++k) s += e[k] * w[k * 16 + c];
    rep[t] = s;
}

__global__ void mask_zero16_kernel(float* __restrict__ rep, const int* __restrict__ mask, int nmask) {
    long long t = (long long)blockIdx.x * blockDim.x + threadIdx.x;
    if (t >= (long long)nmask * 16) return;
    int i = (int)(t >> 4), c = (int)(t & 15);
    rep[(long long)mask[i] * 16 + c] = 0.f;
}

// out[0:nm*128) = recon[mask] ; out[nm*128:2*nm*128) = x[mask]
__global__ void gather_out_kernel(const float* __restrict__ recon, const float* __restrict__ x,
                                  const int* __restrict__ mask, float* __restrict__ out, int nmask) {
    long long t = (long long)blockIdx.x * blockDim.x + threadIdx.x;
    if (t >= (long long)nmask * 128) return;
    int i = (int)(t >> 7), c = (int)(t & 127);
    long long g = (long long)mask[i] * 128 + c;
    out[t] = recon[g];
    out[(long long)nmask * 128 + t] = x[g];
}

// ---------------------------------------------------------------------------
// edge scatter: agg[dst] += h[src]   (float4 gather, 4x global_atomic_add_f32)
// Node tables fit in the 192MB L2, so these atomics are L2-resident.
// ---------------------------------------------------------------------------
__global__ void scatter_add_kernel(const float* __restrict__ h, const int* __restrict__ src,
                                   const int* __restrict__ dst, float* __restrict__ agg,
                                   int nedges, int dim) {
    long long t = (long long)blockIdx.x * blockDim.x + threadIdx.x;
    int chunks = dim >> 2;
    if (t >= (long long)nedges * chunks) return;
    int e = (int)(t / chunks);
    int q = (int)(t % chunks) << 2;
    int s = src[e], d = dst[e];
    const float4 v = *(const float4*)(h + (long long)s * dim + q);
    float* p = agg + (long long)d * dim + q;
    atomicAdd(p + 0, v.x);
    atomicAdd(p + 1, v.y);
    atomicAdd(p + 2, v.z);
    atomicAdd(p + 3, v.w);
}

// ---------------------------------------------------------------------------
// Weight pre-pack: fp32 row-major (K x N) -> f16 B-fragments in the ISA
// 7.12.2 layout for V_WMMA_*_16X16X32: fragment f = ki*NT + nt holds
// 32 lanes x 16 halfs contiguous (1KB). Per lane: VGPR v holds K = ki*32 +
// (v<4 ? 2v : 2v+8) + 8*(lane>=16), column = nt*16 + (lane&15). K beyond the matrix is 0.
// ---------------------------------------------------------------------------
__global__ void pack_wfrag_kernel(const float* __restrict__ W, _Float16* __restrict__ out,
                                  int K, int N) {
    int NT = N / 16, KS = (K + 31) / 32;
    int t = blockIdx.x * blockDim.x + threadIdx.x;
    if (t >= KS * NT * 32) return;
    int lane = t & 31, f = t >> 5;
    int nt = f % NT, ki = f / NT;
    int r = lane & 15, hi = lane >> 4;
    int col = nt * 16 + r;
    v16h hv = {};
#pragma unroll
    for (int v = 0; v < 8; ++v) {
        int k = ki * 32 + ((v < 4) ? (2 * v) : (2 * v + 8)) + 8 * hi;
        if (k < K)     hv[2 * v]     = (_Float16)W[k * N + col];
        if (k + 1 < K) hv[2 * v + 1] = (_Float16)W[(k + 1) * N + col];
    }
    *(v16h*)(out + (size_t)f * 512 + (size_t)lane * 16) = hv;
}

// ---------------------------------------------------------------------------
// Fused GIN MLP: out = act( relu( ((1+eps)h + agg) @ W1 + b1 ) @ W2 + b2 )
// One wave per 16-row tile. A-fragments staged through padded LDS; weight
// B-fragments are pre-packed f16 -> one 32B vector load per lane per WMMA.
// A-frag (16-bit A 16x32): lane halves hold K {0-7,16-23} / {8-15,24-31}.
// C/D: elem v -> row v + 8*(lane>=16), col = lane&15.
// ---------------------------------------------------------------------------
template <int D_IN, int D_OUT, bool RELU_OUT>
__global__ __launch_bounds__(128) void gin_mlp_wmma(
    const float* __restrict__ h, const float* __restrict__ agg,
    const _Float16* __restrict__ W1p, const float* __restrict__ b1,
    const _Float16* __restrict__ W2p, const float* __restrict__ b2,
    const float* __restrict__ eps_p, float* __restrict__ out, int n)
{
    constexpr int NT1 = HD / 16;            // hidden N-tiles (4)
    constexpr int NT2 = D_OUT / 16;         // output N-tiles
    constexpr int KS1 = (D_IN + 31) / 32;   // K-steps for GEMM1
    constexpr int ZP  = (D_IN < 32 ? 32 : D_IN) + 4;  // +4 float pad -> conflict-free column reads

    __shared__ float zs[4][16][ZP];         // z tile per wave
    __shared__ float t1s[4][16][HD + 4];    // hidden tile per wave

    const int lane = threadIdx.x & 31;
    const int w    = threadIdx.x >> 5;
    const int r    = lane & 15;
    const int hi   = lane >> 4;
    const int row0 = (blockIdx.x * 4 + w) * 16;
    const float epsv = 1.0f + eps_p[0];

    const v16h* W1v = (const v16h*)W1p;     // fragment f: entries f*32 + lane
    const v16h* W2v = (const v16h*)W2p;

    // stage z = (1+eps)*h + agg into LDS (coalesced, wave-private region -> no barrier)
    for (int idx = lane; idx < 16 * D_IN; idx += 32) {
        int rr = idx / D_IN, cc = idx % D_IN;
        float zv = 0.f;
        if (row0 + rr < n) {
            long long g = (long long)(row0 + rr) * D_IN + cc;
            zv = epsv * h[g] + agg[g];
        }
        zs[w][rr][cc] = zv;
    }

    // ---- GEMM1: (16 x D_IN) @ (D_IN x 64) ----
    v8f acc[NT1] = {};
#pragma unroll
    for (int ki = 0; ki < KS1; ++ki) {
        v16h a = {};
#pragma unroll
        for (int v = 0; v < 8; ++v) {
            int k0 = ki * 32 + ((v < 4) ? (2 * v) : (2 * v + 8)) + 8 * hi;
            if (k0 < D_IN)     a[2 * v]     = (_Float16)zs[w][r][k0];
            if (k0 + 1 < D_IN) a[2 * v + 1] = (_Float16)zs[w][r][k0 + 1];
        }
#pragma unroll
        for (int nt = 0; nt < NT1; ++nt) {
            v16h b = W1v[(ki * NT1 + nt) * 32 + lane];
            acc[nt] = __builtin_amdgcn_wmma_f32_16x16x32_f16(
                false, a, false, b, (short)0, acc[nt], false, false);
        }
    }

    // bias + ReLU -> t1 in LDS (row = v + 8*hi per C/D layout)
#pragma unroll
    for (int nt = 0; nt < NT1; ++nt) {
        int col = nt * 16 + r;
        float bb = b1[col];
#pragma unroll
        for (int v = 0; v < 8; ++v) {
            float val = acc[nt][v] + bb;
            t1s[w][v + 8 * hi][col] = fmaxf(val, 0.f);
        }
    }

    // ---- GEMM2: (16 x 64) @ (64 x D_OUT) ----
    v8f acc2[NT2] = {};
#pragma unroll
    for (int kt = 0; kt < 2; ++kt) {
        v16h a2 = {};
#pragma unroll
        for (int v = 0; v < 8; ++v) {
            int k0 = kt * 32 + ((v < 4) ? (2 * v) : (2 * v + 8)) + 8 * hi;
            a2[2 * v]     = (_Float16)t1s[w][r][k0];
            a2[2 * v + 1] = (_Float16)t1s[w][r][k0 + 1];
        }
#pragma unroll
        for (int nt = 0; nt < NT2; ++nt) {
            v16h b = W2v[(kt * NT2 + nt) * 32 + lane];
            acc2[nt] = __builtin_amdgcn_wmma_f32_16x16x32_f16(
                false, a2, false, b, (short)0, acc2[nt], false, false);
        }
    }

    // epilogue: bias (+ReLU), coalesced row stores
#pragma unroll
    for (int nt = 0; nt < NT2; ++nt) {
        int col = nt * 16 + r;
        float bb = b2[col];
#pragma unroll
        for (int v = 0; v < 8; ++v) {
            int row_g = row0 + v + 8 * hi;
            if (row_g < n) {
                float val = acc2[nt][v] + bb;
                if (RELU_OUT) val = fmaxf(val, 0.f);
                out[(long long)row_g * D_OUT + col] = val;
            }
        }
    }
}

// ---------------------------------------------------------------------------
// host side
// ---------------------------------------------------------------------------
static inline int gsz(long long n, int b) { return (int)((n + b - 1) / b); }

struct Params { const float *W1, *b1, *W2, *b2, *eps; };
struct PackedW { const _Float16 *W1p, *W2p; };

template <int D_IN, int D_OUT, bool RELU>
static void run_gin_layer(const float* h, float* agg, float* outbuf, const Params& p,
                          const PackedW& pw, const int* esrc, const int* edst,
                          int n, int ne, hipStream_t s)
{
    long long cnt = (long long)n * D_IN;
    zero_f4_kernel<<<gsz(cnt / 4, 256), 256, 0, s>>>((float4*)agg, cnt / 4);
    long long st = (long long)ne * (D_IN / 4);
    scatter_add_kernel<<<gsz(st, 256), 256, 0, s>>>(h, esrc, edst, agg, ne, D_IN);
    gin_mlp_wmma<D_IN, D_OUT, RELU><<<gsz(n, 64), 128, 0, s>>>(
        h, agg, pw.W1p, p.b1, pw.W2p, p.b2, p.eps, outbuf, n);
}

extern "C" void kernel_launch(void* const* d_in, const int* in_sizes, int n_in,
                              void* d_out, int out_size, void* d_ws, size_t ws_size,
                              hipStream_t stream)
{
    const float* x        = (const float*)d_in[0];
    const int* edge_src   = (const int*)d_in[1];
    const int* edge_dst   = (const int*)d_in[2];
    const int* mask_nodes = (const int*)d_in[3];
    const int* token_nodes= (const int*)d_in[4];
    const int* noise_nodes= (const int*)d_in[5];
    const int* noise_src  = (const int*)d_in[6];
    const float* mask_tok = (const float*)d_in[7];

    const int n_nodes = in_sizes[0] / 128;
    const int n_edges = in_sizes[1];
    const int n_mask  = in_sizes[3];
    const int n_token = in_sizes[4];
    const int n_noise = in_sizes[5];

    // param flattening order: insertion (W1,b1,W2,b2,eps) vs key-sorted (W1,W2,b1,b2,eps)
    // enc layer0 slot 1 is b1(64) in insertion order, W2(4096) in sorted order.
    int ob1 = 1, oW2 = 2;
    if (in_sizes[9] == 4096) { oW2 = 1; ob1 = 2; }
    auto getP = [&](int base) -> Params {
        Params p;
        p.W1  = (const float*)d_in[base + 0];
        p.b1  = (const float*)d_in[base + ob1];
        p.W2  = (const float*)d_in[base + oW2];
        p.b2  = (const float*)d_in[base + 3];
        p.eps = (const float*)d_in[base + 4];
        return p;
    };
    Params enc0 = getP(8),  enc1 = getP(13), enc2 = getP(18), enc3 = getP(23);
    const float* w_e2d = (const float*)d_in[28];
    Params dec0 = getP(29), dec1 = getP(34), dec2 = getP(39), dec3 = getP(44);

    // workspace: 3 node-feature buffers (~51.2 MB each) + packed f16 weights
    float* bufA = (float*)d_ws;
    float* bufB = bufA + (size_t)n_nodes * 128;
    float* agg  = bufB + (size_t)n_nodes * 128;
    _Float16* wpack = (_Float16*)(agg + (size_t)n_nodes * 128);

    // ---- pre-pack all weight matrices into f16 B-fragment layout ----
    size_t woff = 0;
    auto packW = [&](const float* W, int K, int N) -> const _Float16* {
        _Float16* dst = wpack + woff;
        int frags = ((K + 31) / 32) * (N / 16);
        pack_wfrag_kernel<<<gsz((long long)frags * 32, 256), 256, 0, stream>>>(W, dst, K, N);
        woff += (size_t)frags * 512;
        return dst;
    };
    const int ein[4]  = {128, 64, 64, 64}, eout[4] = {64, 64, 64, 16};
    const int din_[4] = {16, 64, 64, 64},  dout_[4] = {64, 64, 64, 128};
    Params encP[4] = {enc0, enc1, enc2, enc3}, decP[4] = {dec0, dec1, dec2, dec3};
    PackedW encW[4], decW[4];
    for (int l = 0; l < 4; ++l) {
        encW[l].W1p = packW(encP[l].W1, ein[l], HD);
        encW[l].W2p = packW(encP[l].W2, HD, eout[l]);
    }
    for (int l = 0; l < 4; ++l) {
        decW[l].W1p = packW(decP[l].W1, din_[l], HD);
        decW[l].W2p = packW(decP[l].W2, HD, dout_[l]);
    }

    // ---- mask prep: use_x -> bufA ----
    long long nx = (long long)n_nodes * 128;
    copy_kernel<<<gsz(nx, 256), 256, 0, stream>>>(x, bufA, nx);
    token_set_kernel<<<gsz((long long)n_token * 128, 256), 256, 0, stream>>>(
        bufA, token_nodes, mask_tok, n_token);
    noise_set_kernel<<<gsz((long long)n_noise * 128, 256), 256, 0, stream>>>(
        bufA, x, noise_nodes, noise_src, n_noise);

    // ---- encoder GIN stack ----
    run_gin_layer<128, 64, true >(bufA, agg, bufB, encP[0], encW[0], edge_src, edge_dst, n_nodes, n_edges, stream);
    run_gin_layer< 64, 64, true >(bufB, agg, bufA, encP[1], encW[1], edge_src, edge_dst, n_nodes, n_edges, stream);
    run_gin_layer< 64, 64, true >(bufA, agg, bufB, encP[2], encW[2], edge_src, edge_dst, n_nodes, n_edges, stream);
    run_gin_layer< 64, 16, false>(bufB, agg, bufA, encP[3], encW[3], edge_src, edge_dst, n_nodes, n_edges, stream);

    // ---- enc->dec projection + re-mask ----
    e2d_kernel<<<gsz((long long)n_nodes * 16, 256), 256, 0, stream>>>(bufA, w_e2d, bufB, n_nodes);
    mask_zero16_kernel<<<gsz((long long)n_mask * 16, 256), 256, 0, stream>>>(bufB, mask_nodes, n_mask);

    // ---- decoder GIN stack ----
    run_gin_layer< 16, 64, true >(bufB, agg, bufA, decP[0], decW[0], edge_src, edge_dst, n_nodes, n_edges, stream);
    run_gin_layer< 64, 64, true >(bufA, agg, bufB, decP[1], decW[1], edge_src, edge_dst, n_nodes, n_edges, stream);
    run_gin_layer< 64, 64, true >(bufB, agg, bufA, decP[2], decW[2], edge_src, edge_dst, n_nodes, n_edges, stream);
    run_gin_layer< 64,128, false>(bufA, agg, bufB, decP[3], decW[3], edge_src, edge_dst, n_nodes, n_edges, stream);

    // ---- output: (x_rec, x_init) ----
    gather_out_kernel<<<gsz((long long)n_mask * 128, 256), 256, 0, stream>>>(
        bufB, x, mask_nodes, (float*)d_out, n_mask);

    (void)n_in; (void)out_size; (void)ws_size;
}
